// SwinVideoUnet_41446434406840
// MI455X (gfx1250) — matrix operand order repaired and checked by reading, not compiled, not run on hardware
//
#include <hip/hip_runtime.h>
#include <hip/hip_bf16.h>

// ---------------------------------------------------------------------------
// Swin-Video transformer block for MI455X (gfx1250), wave32 + WMMA f16.
// All GEMMs (QKV, attn QK^T, attn PV, proj, fc1, fc2) use
// v_wmma_f32_16x16x32_f16 with fp32 accumulation.
// ---------------------------------------------------------------------------

typedef _Float16 v16h __attribute__((ext_vector_type(16)));
typedef _Float16 v8h  __attribute__((ext_vector_type(8)));
typedef float    v8f  __attribute__((ext_vector_type(8)));

union V16 {
    v16h v;
    v8h  h[2];
    uint4 u4[2];
};

__device__ __forceinline__ v8f wmma16(const V16& a, const V16& b, v8f c) {
    return __builtin_amdgcn_wmma_f32_16x16x32_f16(
        false, a.v, false, b.v, (short)0, c, false, false);
}

// Problem constants
#define BATCH 2
#define TT 8
#define HH 56
#define WW2 56
#define EMB 384
#define HEADS 12
#define HD 32
#define NTOK 98            // 2*7*7 tokens per window
#define NPAD 112           // padded to 7 wmma tiles
#define NWIN 512           // B * 256 windows
#define MTOT 50176         // 512*98 == 2*8*56*56 tokens
#define QKVC 1152
#define FFN 1536

// ---------------------------------------------------------------------------
// Weight transpose + f32->f16:  W[K][N] -> Wt[N][K] (f16)
// ---------------------------------------------------------------------------
__global__ __launch_bounds__(256) void wt_kernel(const float* __restrict__ W,
                                                 _Float16* __restrict__ Wt,
                                                 int K, int N) {
    int i = blockIdx.x * 256 + threadIdx.x;
    if (i >= K * N) return;
    int k = i / N, n = i % N;
    Wt[(size_t)n * K + k] = (_Float16)W[i];
}

// ---------------------------------------------------------------------------
// Precompute relative-position bias table expanded to [head][row][112] f32.
// Padded columns (>=98) are set to -1e30 so softmax masking is branch-free.
// ---------------------------------------------------------------------------
__global__ __launch_bounds__(256) void bias_prep(const float* __restrict__ table,
                                                 float* __restrict__ biasM) {
    int i = blockIdx.x * 256 + threadIdx.x;
    if (i >= HEADS * NTOK * NPAD) return;
    int head = i / (NTOK * NPAD);
    int rem = i % (NTOK * NPAD);
    int row = rem / NPAD, col = rem % NPAD;
    float v;
    if (col < NTOK) {
        int t1 = row / 49, rm = row % 49, h1 = rm / 7, w1 = rm % 7;
        int t2 = col / 49, rm2 = col % 49, h2 = rm2 / 7, w2 = rm2 % 7;
        int idx = (t1 - t2 + 1) * 169 + (h1 - h2 + 6) * 13 + (w1 - w2 + 6);
        v = table[idx * HEADS + head];
    } else {
        v = -1e30f;
    }
    biasM[i] = v;
}

// ---------------------------------------------------------------------------
// Precompute shift-mask region labels lbl[256 windows][112 tokens] (u8).
// ---------------------------------------------------------------------------
__global__ __launch_bounds__(256) void lbl_prep(unsigned char* __restrict__ lbl) {
    int i = blockIdx.x * 256 + threadIdx.x;   // 256*112 exact
    int wr = i / NPAD, n = i % NPAD;
    unsigned char v = 0;
    if (n < NTOK) {
        int wt = wr >> 6, wh = (wr >> 3) & 7, wwp = wr & 7;
        int nt = n / 49, rn = n % 49, nh = rn / 7, nw = rn % 7;
        int t = wt * 2 + nt, h = wh * 7 + nh, w = wwp * 7 + nw;  // shifted coords
        int a = t < 6 ? 0 : (t < 7 ? 1 : 2);
        int b = h < 49 ? 0 : (h < 53 ? 1 : 2);
        int c = w < 49 ? 0 : (w < 53 ? 1 : 2);
        v = (unsigned char)(a * 9 + b * 3 + c);
    }
    lbl[i] = v;
}

// ---------------------------------------------------------------------------
// LayerNorm (one wave per token).  WINMAP: output rows are window tokens
// (shift + partition applied on the read side).
// ---------------------------------------------------------------------------
template <bool WINMAP>
__global__ __launch_bounds__(256) void ln_kernel(const float* __restrict__ x,
                                                 const float* __restrict__ g,
                                                 const float* __restrict__ b,
                                                 _Float16* __restrict__ out) {
    const int wid = threadIdx.x >> 5, lane = threadIdx.x & 31;
    const int wn = blockIdx.x * 8 + wid;   // output token id
    size_t src;
    if (WINMAP) {
        int w = wn / NTOK, n = wn % NTOK;
        int bb = w / 256, r = w % 256;
        int wt = r / 64, wh = (r / 8) & 7, wwp = r & 7;
        int nt = n / 49, rn = n % 49, nh = rn / 7, nw = rn % 7;
        int t = wt * 2 + nt, h = wh * 7 + nh, wd = wwp * 7 + nw;  // shifted coords
        int st = (t + 1) & 7, sh = (h + 3) % 56, sw = (wd + 3) % 56;
        src = (((size_t)bb * TT + st) * HH + sh) * WW2 + sw;
    } else {
        src = (size_t)wn;
    }
    const float* px = x + src * EMB;
    float vals[12], s1 = 0.f, s2 = 0.f;
#pragma unroll
    for (int i = 0; i < 12; ++i) {
        float v = px[lane + 32 * i];
        vals[i] = v; s1 += v; s2 += v * v;
    }
#pragma unroll
    for (int off = 16; off; off >>= 1) {
        s1 += __shfl_xor(s1, off);
        s2 += __shfl_xor(s2, off);
    }
    float mean = s1 * (1.f / EMB);
    float var  = s2 * (1.f / EMB) - mean * mean;
    float inv  = rsqrtf(var + 1e-5f);
    _Float16* po = out + (size_t)wn * EMB;
#pragma unroll
    for (int i = 0; i < 12; ++i) {
        int c = lane + 32 * i;
        po[c] = (_Float16)((vals[i] - mean) * inv * g[c] + b[c]);
    }
}

// ---------------------------------------------------------------------------
// Generic WMMA GEMM:  out[M][N] = A[M][K] (f16) * Bt[N][K]^T (f16) + bias
// Tiles: 128x128x32, 256 threads = 8 waves (2x4), each wave 64x32.
// M,N,K all multiples of 128/128/32 for this problem (no bounds checks).
// ---------------------------------------------------------------------------
__device__ __forceinline__ float gelu_exact(float x) {
    return 0.5f * x * (1.0f + erff(x * 0.70710678118654752f));
}

template <bool OUT_F16, bool GELU_ACT, bool RESID>
__global__ __launch_bounds__(256) void gemm_wmma(const _Float16* __restrict__ A,
                                                 const _Float16* __restrict__ Bt,
                                                 const float* __restrict__ bias,
                                                 void* __restrict__ out,
                                                 const float* __restrict__ resid,
                                                 int M, int N, int K) {
    __shared__ _Float16 As[128 * 40];
    __shared__ _Float16 Bs[128 * 40];
    const int tid = threadIdx.x, lane = tid & 31;
    const int wid = tid >> 5;
    const int wm = wid >> 2;   // 0..1 -> 64-row slab
    const int wn = wid & 3;    // 0..3 -> 32-col slab
    const int lg = lane >> 4, lr = lane & 15;
    const int bm = blockIdx.x, bn = blockIdx.y;

    const int r = tid >> 1, c = (tid & 1) * 16;
    const _Float16* gA = A + (size_t)(bm * 128 + r) * K + c;
    const _Float16* gB = Bt + (size_t)(bn * 128 + r) * K + c;

    v8f acc[4][2] = {};

    for (int k0 = 0; k0 < K; k0 += 32) {
        {   // cooperative tile load: each thread moves 32B of A and 32B of B
            const uint4* sa = (const uint4*)(gA + k0);
            uint4 a0 = sa[0], a1 = sa[1];
            *(uint4*)(&As[r * 40 + c])     = a0;
            *(uint4*)(&As[r * 40 + c + 8]) = a1;
            const uint4* sb = (const uint4*)(gB + k0);
            uint4 b0 = sb[0], b1 = sb[1];
            *(uint4*)(&Bs[r * 40 + c])     = b0;
            *(uint4*)(&Bs[r * 40 + c + 8]) = b1;
            if (k0 + 32 < K) {   // prefetch next K-tile (global_prefetch_b8)
                __builtin_prefetch(gA + k0 + 32, 0, 0);
                __builtin_prefetch(gB + k0 + 32, 0, 0);
            }
        }
        __syncthreads();
        V16 a[4], b[2];
#pragma unroll
        for (int mt = 0; mt < 4; ++mt) {
            const _Float16* p = &As[(wm * 64 + mt * 16 + lr) * 40 + lg * 8];
            a[mt].h[0] = *(const v8h*)(p);
            a[mt].h[1] = *(const v8h*)(p + 16);
        }
#pragma unroll
        for (int nt = 0; nt < 2; ++nt) {
            const _Float16* p = &Bs[(wn * 32 + nt * 16 + lr) * 40 + lg * 8];
            b[nt].h[0] = *(const v8h*)(p);
            b[nt].h[1] = *(const v8h*)(p + 16);
        }
#pragma unroll
        for (int mt = 0; mt < 4; ++mt)
#pragma unroll
            for (int nt = 0; nt < 2; ++nt)
                acc[mt][nt] = wmma16(a[mt], b[nt], acc[mt][nt]);
        __syncthreads();
    }

#pragma unroll
    for (int mt = 0; mt < 4; ++mt) {
#pragma unroll
        for (int nt = 0; nt < 2; ++nt) {
            int col = bn * 128 + wn * 32 + nt * 16 + lr;
            float bc = bias[col];
#pragma unroll
            for (int rr = 0; rr < 8; ++rr) {
                int row = bm * 128 + wm * 64 + mt * 16 + rr + lg * 8;
                float v = acc[mt][nt][rr] + bc;
                if (GELU_ACT) v = gelu_exact(v);
                if (RESID)    v += resid[(size_t)row * N + col];
                if (OUT_F16)  ((_Float16*)out)[(size_t)row * N + col] = (_Float16)v;
                else          ((float*)out)[(size_t)row * N + col] = v;
            }
        }
    }
}

// ---------------------------------------------------------------------------
// Windowed attention, one block per (window, head), 128 threads = 4 waves.
// S = (q k^T)*scale + rel_bias + shift_mask ; P = softmax(S) ; O = P v.
// Bias (with -1e30 pad cols) and region labels come from precomputed tables.
// ---------------------------------------------------------------------------
__global__ __launch_bounds__(128) void attn_kernel(const _Float16* __restrict__ qkv,
                                                   const float* __restrict__ biasM,
                                                   const unsigned char* __restrict__ lbl,
                                                   _Float16* __restrict__ o) {
    __shared__ _Float16 Ps[112 * 136];   // softmax probs, K padded to 128
    __shared__ _Float16 Vt[32 * 136];    // V^T, rows >= 98 zeroed

    const int w = blockIdx.x;        // 0..511
    const int head = blockIdx.y;     // 0..11
    const int tid = threadIdx.x, lane = tid & 31, wid = tid >> 5;
    const int lg = lane >> 4, lr = lane & 15;
    const size_t base = (size_t)w * NTOK * QKVC;
    const unsigned char* lblw = lbl + (size_t)(w & 255) * NPAD;

    // ---- stage V^T into LDS (zero pad m in [98,128)) ----
    for (int i = tid; i < 32 * 128; i += 128) {
        int d = i >> 7, m = i & 127;
        _Float16 v = (m < NTOK)
            ? qkv[base + (size_t)m * QKVC + 2 * EMB + head * HD + d]
            : (_Float16)0.f;
        Vt[d * 136 + m] = v;
    }
    // zero Ps pad columns [112,128)
    for (int i = tid; i < 112 * 16; i += 128) {
        int rr = i >> 4, cc = 112 + (i & 15);
        Ps[rr * 136 + cc] = (_Float16)0.f;
    }

    // ---- S tiles + softmax, per wave: row-tiles {wid, wid+4} ----
    for (int rt = wid; rt < 7; rt += 4) {
        V16 aq;
        {
            int row = rt * 16 + lr;
            int tok = row < NTOK ? row : NTOK - 1;
            const _Float16* p = qkv + base + (size_t)tok * QKVC + head * HD + lg * 8;
            aq.u4[0] = *(const uint4*)(p);
            aq.u4[1] = *(const uint4*)(p + 16);
        }
        v8f s[7];
        int l2[7];
#pragma unroll
        for (int ct = 0; ct < 7; ++ct) {
            V16 bk;
            int mcol = ct * 16 + lr;
            int tok = mcol < NTOK ? mcol : NTOK - 1;
            const _Float16* p = qkv + base + (size_t)tok * QKVC + EMB + head * HD + lg * 8;
            bk.u4[0] = *(const uint4*)(p);
            bk.u4[1] = *(const uint4*)(p + 16);
            v8f z = {};
            s[ct] = wmma16(aq, bk, z);
            l2[ct] = lblw[mcol];
        }
        // softmax rows: VGPR slot r holds row rt*16 + r + lg*8, col = ct*16 + lr
#pragma unroll
        for (int r = 0; r < 8; ++r) {
            int row = rt * 16 + r + lg * 8;
            int rowc = row < NTOK ? row : NTOK - 1;
            int lbl1 = lblw[rowc];
            const float* brow = biasM + ((size_t)head * NTOK + rowc) * NPAD;
            float tmp[7], mx = -3e38f;
#pragma unroll
            for (int ct = 0; ct < 7; ++ct) {
                float val = fmaf(s[ct][r], 0.17677669529663689f,
                                 brow[ct * 16 + lr]) +
                            (lbl1 != l2[ct] ? -100.f : 0.f);
                tmp[ct] = val;
                mx = fmaxf(mx, val);
            }
#pragma unroll
            for (int off = 8; off; off >>= 1) mx = fmaxf(mx, __shfl_xor(mx, off));
            float sum = 0.f;
#pragma unroll
            for (int ct = 0; ct < 7; ++ct) {
                float e = __expf(tmp[ct] - mx);
                tmp[ct] = e; sum += e;
            }
#pragma unroll
            for (int off = 8; off; off >>= 1) sum += __shfl_xor(sum, off);
            float inv = 1.f / fmaxf(sum, 1e-20f);
#pragma unroll
            for (int ct = 0; ct < 7; ++ct)
                Ps[row * 136 + ct * 16 + lr] = (_Float16)(tmp[ct] * inv);
        }
    }
    __syncthreads();

    // ---- O = P @ V  (K = 128 padded, 4 wmma K-steps) ----
    for (int rt = wid; rt < 7; rt += 4) {
        v8f oacc[2] = {};
#pragma unroll
        for (int ks = 0; ks < 4; ++ks) {
            V16 ap;
            const _Float16* p = &Ps[(rt * 16 + lr) * 136 + ks * 32 + lg * 8];
            ap.h[0] = *(const v8h*)(p);
            ap.h[1] = *(const v8h*)(p + 16);
#pragma unroll
            for (int nt = 0; nt < 2; ++nt) {
                V16 bv;
                const _Float16* pv = &Vt[(nt * 16 + lr) * 136 + ks * 32 + lg * 8];
                bv.h[0] = *(const v8h*)(pv);
                bv.h[1] = *(const v8h*)(pv + 16);
                oacc[nt] = wmma16(ap, bv, oacc[nt]);
            }
        }
#pragma unroll
        for (int nt = 0; nt < 2; ++nt)
#pragma unroll
            for (int r = 0; r < 8; ++r) {
                int row = rt * 16 + r + lg * 8;
                if (row < NTOK)
                    o[((size_t)w * NTOK + row) * EMB + head * HD + nt * 16 + lr] =
                        (_Float16)oacc[nt][r];
            }
    }
}

// ---------------------------------------------------------------------------
// Un-partition (window reverse + roll back) + first residual:
//   x1[nat] = x[nat] + proj_out[window(nat)]
// ---------------------------------------------------------------------------
__global__ __launch_bounds__(256) void resid_kernel(const float* __restrict__ x,
                                                    const float* __restrict__ proj,
                                                    float* __restrict__ x1) {
    size_t i = (size_t)blockIdx.x * 256 + threadIdx.x;   // over MTOT*EMB (exact)
    size_t tn = i / EMB;
    int c = (int)(i % EMB);
    int bb = (int)(tn / 25088);
    int r2 = (int)(tn % 25088);
    int t = r2 / 3136, r3 = r2 % 3136, h = r3 / 56, wd = r3 % 56;
    int ts = (t + 7) & 7, hs = (h + 53) % 56, ws2 = (wd + 53) % 56;
    int wt = ts >> 1, nt = ts & 1;
    int whp = hs / 7, nh = hs % 7, wwp = ws2 / 7, nw = ws2 % 7;
    int w = ((bb * 4 + wt) * 8 + whp) * 8 + wwp;
    int n = (nt * 7 + nh) * 7 + nw;
    x1[i] = x[i] + proj[((size_t)w * NTOK + n) * EMB + c];
}

// ---------------------------------------------------------------------------
// Launcher
// ---------------------------------------------------------------------------
extern "C" void kernel_launch(void* const* d_in, const int* in_sizes, int n_in,
                              void* d_out, int out_size, void* d_ws, size_t ws_size,
                              hipStream_t stream) {
    const float* x          = (const float*)d_in[0];
    const float* g1         = (const float*)d_in[1];
    const float* b1         = (const float*)d_in[2];
    const float* qkv_w      = (const float*)d_in[3];
    const float* qkv_b      = (const float*)d_in[4];
    const float* bias_table = (const float*)d_in[5];
    const float* proj_w     = (const float*)d_in[6];
    const float* proj_b     = (const float*)d_in[7];
    const float* g2         = (const float*)d_in[8];
    const float* b2         = (const float*)d_in[9];
    const float* fc1_w      = (const float*)d_in[10];
    const float* fc1_b      = (const float*)d_in[11];
    const float* fc2_w      = (const float*)d_in[12];
    const float* fc2_b      = (const float*)d_in[13];

    char* ws = (char*)d_ws;
    // workspace layout (aliasing exploits dead ranges; ~320 MB total)
    const size_t OFF_XW  = 0;                        // xw f16 (38.5MB); later o f16; later fc1_out
    const size_t OFF_QKV = 40ull  * 1024 * 1024;     // qkv f16 (115.6MB); later proj_out f32 (77MB)
    const size_t OFF_X1  = 160ull * 1024 * 1024;     // x1 f32 (77MB)
    const size_t OFF_H   = 248ull * 1024 * 1024;     // h f16 (38.5MB)
    const size_t OFF_W   = 300ull * 1024 * 1024;     // transposed f16 weights
    const size_t OFF_B   = 305ull * 1024 * 1024;     // biasM f32 (527KB) + lbl u8 (28KB)

    _Float16* xw      = (_Float16*)(ws + OFF_XW);
    _Float16* qkvbuf  = (_Float16*)(ws + OFF_QKV);
    _Float16* o16     = (_Float16*)(ws + OFF_XW);    // alias: xw dead after QKV GEMM
    float*    projout = (float*)   (ws + OFF_QKV);   // alias: qkv dead after attention
    float*    x1      = (float*)   (ws + OFF_X1);
    _Float16* h16     = (_Float16*)(ws + OFF_H);
    _Float16* fc1o    = (_Float16*)(ws + OFF_XW);    // alias: o + projout dead by fc1

    _Float16* qkv_wT  = (_Float16*)(ws + OFF_W);
    _Float16* proj_wT = qkv_wT + (size_t)QKVC * EMB;
    _Float16* fc1_wT  = proj_wT + (size_t)EMB * EMB;
    _Float16* fc2_wT  = fc1_wT + (size_t)FFN * EMB;
    float*    biasM   = (float*)(ws + OFF_B);
    unsigned char* lbl = (unsigned char*)(biasM + (size_t)HEADS * NTOK * NPAD);

    // 1) transpose weights to f16 [N][K]; expand bias + labels
    wt_kernel<<<(EMB * QKVC + 255) / 256, 256, 0, stream>>>(qkv_w, qkv_wT, EMB, QKVC);
    wt_kernel<<<(EMB * EMB + 255) / 256, 256, 0, stream>>>(proj_w, proj_wT, EMB, EMB);
    wt_kernel<<<(EMB * FFN + 255) / 256, 256, 0, stream>>>(fc1_w, fc1_wT, EMB, FFN);
    wt_kernel<<<(FFN * EMB + 255) / 256, 256, 0, stream>>>(fc2_w, fc2_wT, FFN, EMB);
    bias_prep<<<(HEADS * NTOK * NPAD + 255) / 256, 256, 0, stream>>>(bias_table, biasM);
    lbl_prep<<<(256 * NPAD) / 256, 256, 0, stream>>>(lbl);

    // 2) LN1 + cyclic shift + window partition -> xw (window-token order, f16)
    ln_kernel<true><<<MTOT / 8, 256, 0, stream>>>(x, g1, b1, xw);

    // 3) QKV GEMM: (50176 x 384) x (384 x 1152)
    gemm_wmma<true, false, false><<<dim3(MTOT / 128, QKVC / 128), 256, 0, stream>>>(
        xw, qkv_wT, qkv_b, qkvbuf, nullptr, MTOT, QKVC, EMB);

    // 4) windowed attention (512 windows x 12 heads)
    attn_kernel<<<dim3(NWIN, HEADS), 128, 0, stream>>>(qkvbuf, biasM, lbl, o16);

    // 5) proj GEMM: (50176 x 384) x (384 x 384) -> f32 (window order)
    gemm_wmma<false, false, false><<<dim3(MTOT / 128, EMB / 128), 256, 0, stream>>>(
        o16, proj_wT, proj_b, projout, nullptr, MTOT, EMB, EMB);

    // 6) window reverse + roll + residual -> x1 (natural order, f32)
    resid_kernel<<<(MTOT * EMB) / 256, 256, 0, stream>>>(x, projout, x1);

    // 7) LN2 -> h (natural order, f16)
    ln_kernel<false><<<MTOT / 8, 256, 0, stream>>>(x1, g2, b2, h16);

    // 8) fc1 GEMM + exact GELU: (50176 x 384) x (384 x 1536)
    gemm_wmma<true, true, false><<<dim3(MTOT / 128, FFN / 128), 256, 0, stream>>>(
        h16, fc1_wT, fc1_b, fc1o, nullptr, MTOT, FFN, EMB);

    // 9) fc2 GEMM + fused residual(x1) -> d_out (f32)
    gemm_wmma<false, false, true><<<dim3(MTOT / 128, EMB / 128), 256, 0, stream>>>(
        fc1o, fc2_wT, fc2_b, d_out, x1, MTOT, EMB, FFN);
}
